// MatchNet_5351529250968
// MI455X (gfx1250) — matrix operand order repaired and checked by reading, not compiled, not run on hardware
//
#include <hip/hip_runtime.h>

// ---------------------------------------------------------------------------
// Types
// ---------------------------------------------------------------------------
typedef __bf16 bf16_t;
typedef __attribute__((ext_vector_type(16))) __bf16 v16bf;
typedef __attribute__((ext_vector_type(8)))  float  v8f;
typedef __attribute__((ext_vector_type(4)))  float  v4f;
typedef __attribute__((ext_vector_type(2)))  float  v2f;
typedef __attribute__((ext_vector_type(4)))  unsigned int v4u;
typedef __attribute__((ext_vector_type(8)))  int    v8i;
typedef __attribute__((ext_vector_type(4)))  int    v4i;

union BF16Frag { v16bf v; v4f f4[2]; };

// Problem constants (match reference)
#define NB      2048
#define NCOMBO  64
#define NSTRUCT 512
#define HID     1024
#define NITER   60
#define TK      32

// ---------------------------------------------------------------------------
// TDM availability (ROCm7.2 clang-22: 5-arg builtin; clang-23: 6-arg)
// ---------------------------------------------------------------------------
#if defined(__has_builtin)
#if __has_builtin(__builtin_amdgcn_tensor_load_to_lds) && \
    __has_builtin(__builtin_amdgcn_s_wait_tensorcnt)
#define USE_TDM 1
#endif
#endif
#ifndef USE_TDM
#define USE_TDM 0
#endif

#if USE_TDM
// Issue a TDM load of a 2-D bf16 tile: 128 rows x 32 elems, row stride
// `row_stride_elems`, into LDS at byte offset `lds_addr`, inserting 4 dwords
// of padding after every 16 dwords (one 64-byte row) -> LDS rows of 40 bf16.
__device__ __forceinline__ void tdm_load_tile_128x32_bf16(
    unsigned lds_addr, const bf16_t* gptr, int row_stride_elems)
{
    unsigned long long ga = (unsigned long long)gptr;
    // D# group 0: count=1 | lds_addr | global_addr[56:0] | type=2
    v4u g0 = { 1u,
               lds_addr,
               (unsigned)(ga & 0xFFFFFFFFu),
               (unsigned)((ga >> 32) & 0x01FFFFFFu) | (2u << 30) };
    // D# group 1
    v8i g1;
    g1[0] = (1 << 16)        // data_size = 2 bytes
          | (1 << 20)        // pad_enable
          | (3 << 22)        // pad_interval: every 16 dwords
          | (3 << 25);       // pad_amount: 4 dwords
    g1[1] = (TK & 0xFFFF) << 16;                         // tensor_dim0 lo16
    g1[2] = ((TK >> 16) & 0xFFFF) | (128 << 16);         // dim0 hi | dim1 lo
    g1[3] = (TK << 16);                                  // dim1 hi | tile_dim0
    g1[4] = 128;                                         // tile_dim1 | tile_dim2=0
    g1[5] = row_stride_elems;                            // dim0_stride lo32
    g1[6] = 0;                                           // dim0_stride hi | dim1_stride lo
    g1[7] = 0;                                           // dim1_stride hi
    v4i gz = { 0, 0, 0, 0 };
#if __clang_major__ >= 23
    v8i gz8 = { 0, 0, 0, 0, 0, 0, 0, 0 };
    __builtin_amdgcn_tensor_load_to_lds(g0, g1, gz, gz, gz8, 0);
#else
    __builtin_amdgcn_tensor_load_to_lds(g0, g1, gz, gz, 0);
#endif
}
#endif

// Low 32 bits of a flat pointer to LDS == LDS byte offset (ISA 10.2 aperture).
__device__ __forceinline__ unsigned lds_addr_of(const void* p) {
    return (unsigned)(unsigned long long)p;
}

// ---------------------------------------------------------------------------
// Kernel 1: power iteration on S^T S + I  ->  tau = 0.9 / ||K||_2
// ---------------------------------------------------------------------------
__global__ __launch_bounds__(256) void power_iter_kernel(
    const float* __restrict__ Sg, float* __restrict__ tau_out)
{
    __shared__ float v[NSTRUCT];
    __shared__ float w[NSTRUCT];
    __shared__ float sv[NCOMBO];
    __shared__ float red[256];
    const int tid = threadIdx.x;

    for (int i = tid; i < NSTRUCT; i += 256) v[i] = 0.04419417382415922f; // 1/sqrt(512)
    __syncthreads();

    for (int iter = 0; iter < 30; ++iter) {
        if (tid < NCOMBO) {
            float s = 0.f;
            for (int k = 0; k < NSTRUCT; ++k) s += Sg[tid * NSTRUCT + k] * v[k];
            sv[tid] = s;
        }
        __syncthreads();
        float part = 0.f;
        for (int i = tid; i < NSTRUCT; i += 256) {
            float s = v[i];
            for (int r = 0; r < NCOMBO; ++r) s += Sg[r * NSTRUCT + i] * sv[r];
            w[i] = s;
            part += s * s;
        }
        red[tid] = part;
        __syncthreads();
        for (int s = 128; s > 0; s >>= 1) {
            if (tid < s) red[tid] += red[tid + s];
            __syncthreads();
        }
        const float nrm = sqrtf(red[0]);
        __syncthreads();
        for (int i = tid; i < NSTRUCT; i += 256) v[i] = w[i] / nrm;
        __syncthreads();
    }
    if (tid < NCOMBO) {
        float s = 0.f;
        for (int k = 0; k < NSTRUCT; ++k) s += Sg[tid * NSTRUCT + k] * v[k];
        sv[tid] = s;
    }
    __syncthreads();
    float part = 0.f;
    for (int i = tid; i < NSTRUCT; i += 256) {
        float s = v[i];
        for (int r = 0; r < NCOMBO; ++r) s += Sg[r * NSTRUCT + i] * sv[r];
        part += s * v[i];
    }
    red[tid] = part;
    __syncthreads();
    for (int s = 128; s > 0; s >>= 1) {
        if (tid < s) red[tid] += red[tid + s];
        __syncthreads();
    }
    if (tid == 0) tau_out[0] = 0.9f / sqrtf(red[0]);
}

// ---------------------------------------------------------------------------
// Kernel 2a: f32 -> bf16 conversion (for X)
// ---------------------------------------------------------------------------
__global__ __launch_bounds__(256) void convert_f32_bf16(
    const float* __restrict__ src, bf16_t* __restrict__ dst, int n)
{
    int i = blockIdx.x * 256 + threadIdx.x;
    if (i < n) dst[i] = (bf16_t)src[i];
}

// ---------------------------------------------------------------------------
// Kernel 2b: W [K,N] f32 -> Wt [N,K] bf16 (32x32 LDS-tiled transpose)
// ---------------------------------------------------------------------------
__global__ __launch_bounds__(256) void transpose_to_bf16(
    const float* __restrict__ W, bf16_t* __restrict__ Wt, int K, int N)
{
    __shared__ float t[32][33];
    const int kb = blockIdx.y * 32, nb = blockIdx.x * 32;
    const int tx = threadIdx.x & 31, ty = threadIdx.x >> 5;  // 8 rows per pass
#pragma unroll
    for (int i = 0; i < 32; i += 8)
        t[ty + i][tx] = W[(size_t)(kb + ty + i) * N + nb + tx];
    __syncthreads();
#pragma unroll
    for (int i = 0; i < 32; i += 8)
        Wt[(size_t)(nb + ty + i) * K + kb + tx] = (bf16_t)t[tx][ty + i];
}

// ---------------------------------------------------------------------------
// Kernel 3: bf16 WMMA GEMM  C = relu(A @ Bt^T + bias)
//   A  [M,K] bf16 row-major (activations)
//   Bt [N,K] bf16 row-major (pre-transposed weights)
//   Block tile 128x128, k-step 32, 8 waves, wave tile 32x64.
//   Tiles streamed global->LDS by the Tensor Data Mover, double-buffered.
// ---------------------------------------------------------------------------
template <bool OUT_BF16>
__global__ __launch_bounds__(256) void gemm_bias_relu(
    const bf16_t* __restrict__ A, const bf16_t* __restrict__ Bt,
    const float* __restrict__ bias, bf16_t* __restrict__ outB,
    float* __restrict__ outF, int M, int N, int K)
{
    __shared__ bf16_t As[2][128][TK + 8];   // [row][k], TDM-padded rows
    __shared__ bf16_t Bs[2][128][TK + 8];   // [n][k]

    const int tid  = threadIdx.x;
    const int lane = tid & 31;
    const int wave = tid >> 5;
    const int wm   = wave & 3;    // 4 M-groups of 32
    const int wn   = wave >> 2;   // 2 N-groups of 64
    const int bm   = blockIdx.y * 128;
    const int bn   = blockIdx.x * 128;

    v8f acc[2][4] = {};

    const int mr    = lane & 15;
    const int halfA = (lane >> 4) * 8;    // 16-bit A frag K sub-offset
    const int halfB = (lane >> 4) * 8;    // 16-bit B frag K sub-offset
    const int nsteps = K / TK;

    auto issue_tiles = [&](int buf, int k0) {
#if USE_TDM
        if (wave == 0) {
            tdm_load_tile_128x32_bf16(lds_addr_of(&As[buf][0][0]),
                                      A + (size_t)bm * K + k0, K);
            tdm_load_tile_128x32_bf16(lds_addr_of(&Bs[buf][0][0]),
                                      Bt + (size_t)bn * K + k0, K);
        }
#else
        const int row = tid >> 1;             // 0..127
        const int off = (tid & 1) * 16;       // 0 / 16
        const bf16_t* sa = A  + (size_t)(bm + row) * K + k0 + off;
        const bf16_t* sb = Bt + (size_t)(bn + row) * K + k0 + off;
        *reinterpret_cast<v4f*>(&As[buf][row][off])     = *reinterpret_cast<const v4f*>(sa);
        *reinterpret_cast<v4f*>(&As[buf][row][off + 8]) = *reinterpret_cast<const v4f*>(sa + 8);
        *reinterpret_cast<v4f*>(&Bs[buf][row][off])     = *reinterpret_cast<const v4f*>(sb);
        *reinterpret_cast<v4f*>(&Bs[buf][row][off + 8]) = *reinterpret_cast<const v4f*>(sb + 8);
        if (k0 + TK < K) {                    // prefetch next tiles
            __builtin_prefetch(sa + TK, 0, 1);
            __builtin_prefetch(sb + TK, 0, 1);
        }
#endif
    };

    issue_tiles(0, 0);
    for (int s = 0; s < nsteps; ++s) {
        const int buf = s & 1;
        if (s + 1 < nsteps) issue_tiles(buf ^ 1, (s + 1) * TK);
#if USE_TDM
        if (wave == 0) {
            if (s + 1 < nsteps) __builtin_amdgcn_s_wait_tensorcnt(2);  // next 2 still in flight
            else                __builtin_amdgcn_s_wait_tensorcnt(0);  // drain
        }
#endif
        __syncthreads();

        BF16Frag aF[2];
#pragma unroll
        for (int mi = 0; mi < 2; ++mi) {
            const bf16_t* p = &As[buf][wm * 32 + mi * 16 + mr][0];
            aF[mi].f4[0] = *reinterpret_cast<const v4f*>(p + halfA);      // K 0..7 / 8..15
            aF[mi].f4[1] = *reinterpret_cast<const v4f*>(p + halfA + 16); // K 16..23 / 24..31
        }
        BF16Frag bF[4];
#pragma unroll
        for (int ni = 0; ni < 4; ++ni) {
            const bf16_t* p = &Bs[buf][wn * 64 + ni * 16 + mr][halfB];
            bF[ni].f4[0] = *reinterpret_cast<const v4f*>(p);
            bF[ni].f4[1] = *reinterpret_cast<const v4f*>(p + 16);
        }
#pragma unroll
        for (int mi = 0; mi < 2; ++mi)
#pragma unroll
            for (int ni = 0; ni < 4; ++ni)
                acc[mi][ni] = __builtin_amdgcn_wmma_f32_16x16x32_bf16(
                    false, aF[mi].v, false, bF[ni].v, (short)0, acc[mi][ni], false, false);
        __syncthreads();
    }

    // Epilogue: bias + relu. C/D layout: VGPR r -> (M = r + 8*(lane>=16), N = lane%16)
    const int rowHalf = (lane >> 4) * 8;
#pragma unroll
    for (int mi = 0; mi < 2; ++mi) {
#pragma unroll
        for (int ni = 0; ni < 4; ++ni) {
            const int col = bn + wn * 64 + ni * 16 + mr;
            const float bsv = bias[col];
#pragma unroll
            for (int r = 0; r < 8; ++r) {
                const int row = bm + wm * 32 + mi * 16 + rowHalf + r;
                float v = acc[mi][ni][r] + bsv;
                v = v > 0.f ? v : 0.f;
                if (OUT_BF16) outB[(size_t)row * N + col] = (bf16_t)v;
                else          outF[(size_t)row * N + col] = v;
            }
        }
    }
}

// ---------------------------------------------------------------------------
// Kernel 4: fused PDHG solver. One block = 16 batch rows, all state in LDS.
//   f32 WMMA (16x16x4) for both S-matmuls -> exact fp32 products with 0/1 S.
// ---------------------------------------------------------------------------
#define SSTR 516   // padded stride for 512-wide rows
#define CSTR 68    // padded stride for 64-wide rows

// S 64*516 | x,xbV,Z,y2 4*16*516 | y1,Bc,U,U2 4*16*68 | red 16*17 | scale 16
#define PDHG_LDS_FLOATS (64*SSTR + 4*16*SSTR + 4*16*CSTR + 16*17 + 16)
#define PDHG_LDS_BYTES  (PDHG_LDS_FLOATS * 4)

__global__ __launch_bounds__(256) void pdhg_kernel(
    const float* __restrict__ Sg, const float* __restrict__ Zg,
    const float* __restrict__ Xg, const float* __restrict__ tau_ptr,
    float* __restrict__ outX)
{
    extern __shared__ float smem[];
    float* S   = smem;                 // [64][516]
    float* x   = S   + 64 * SSTR;      // [16][516]
    float* xbV = x   + 16 * SSTR;      // [16][516]  xbar -> V -> d -> new xbar
    float* Z   = xbV + 16 * SSTR;      // [16][516]
    float* y2  = Z   + 16 * SSTR;      // [16][516]
    float* y1  = y2  + 16 * SSTR;      // [16][68]
    float* Bc  = y1  + 16 * CSTR;      // [16][68]
    float* U   = Bc  + 16 * CSTR;      // [16][68]  K 0..255 partial
    float* U2  = U   + 16 * CSTR;      // [16][68]  K 256..511 partial
    float* red = U2  + 16 * CSTR;      // [16][17]
    float* scl = red + 16 * 17;        // [16]

    const int tid  = threadIdx.x;
    const int lane = tid & 31;
    const int wave = tid >> 5;
    const int mr   = lane & 15;
    const int kh   = (lane >> 4) * 2;      // f32 WMMA: lanes>=16 hold K+2,K+3
    const int rowHalf = (lane >> 4) * 8;   // C/D layout row half
    const int r0   = blockIdx.x * 16;

    const float tau   = tau_ptr[0];
    const float sigma = tau;

    // ---- init ----
    for (int e = tid; e < NCOMBO * NSTRUCT; e += 256) {
        int r = e >> 9, c = e & 511;
        S[r * SSTR + c] = Sg[e];
    }
    for (int e = tid; e < 16 * NSTRUCT; e += 256) {
        int r = e >> 9, c = e & 511;
        Z[r * SSTR + c]   = Zg[(size_t)(r0 + r) * NSTRUCT + c];
        x[r * SSTR + c]   = 0.f;
        xbV[r * SSTR + c] = 0.f;
        y2[r * SSTR + c]  = 0.f;
    }
    for (int e = tid; e < 16 * NCOMBO; e += 256) {
        int r = e >> 6, c = e & 63;
        Bc[r * CSTR + c] = Xg[(size_t)(r0 + r) * NCOMBO + c];
        y1[r * CSTR + c] = 0.f;
        U[r * CSTR + c]  = 0.f;
        U2[r * CSTR + c] = 0.f;
    }
    __syncthreads();

    const int drow = tid >> 4;   // 0..15  (norm pass: 16 threads / row)
    const int dcl  = tid & 15;

    for (int it = 0; it < NITER; ++it) {
        // ---- Phase A: U+U2 = xbar @ S^T (16x64, K=512), all 8 waves ----
        {
            v8f acc = {};
            const int half = wave >> 2;              // K half
            const int nt   = (wave & 3) * 16;        // N tile base
            const int n    = nt + mr;
            const int kbeg = half * 256;
            for (int k = kbeg; k < kbeg + 256; k += 4) {
                v2f a, b;
                a[0] = xbV[mr * SSTR + k + kh];
                a[1] = xbV[mr * SSTR + k + kh + 1];
                b[0] = S[n * SSTR + k + kh];         // B[k][n] = S[n][k]
                b[1] = S[n * SSTR + k + kh + 1];
                acc = __builtin_amdgcn_wmma_f32_16x16x4_f32(
                    false, a, false, b, (short)0, acc, false, false);
            }
            float* Ut = half ? U2 : U;
#pragma unroll
            for (int r = 0; r < 8; ++r)
                Ut[(rowHalf + r) * CSTR + n] = acc[r];
        }
        __syncthreads();

        // ---- Phase B: dual update y = relu(y + sigma*(Kx(xbar) - c)) ----
        for (int e = tid; e < 16 * NCOMBO; e += 256) {
            int r = e >> 6, c = e & 63;
            float u = U[r * CSTR + c] + U2[r * CSTR + c];
            float v = y1[r * CSTR + c] + sigma * (u - Bc[r * CSTR + c]);
            y1[r * CSTR + c] = v > 0.f ? v : 0.f;
        }
        for (int e = tid; e < 16 * NSTRUCT; e += 256) {
            int r = e >> 9, c = e & 511;
            float v = y2[r * SSTR + c] - sigma * xbV[r * SSTR + c];
            y2[r * SSTR + c] = v > 0.f ? v : 0.f;
        }
        __syncthreads();   // xbar dead -> buffer becomes V

        // ---- Phase C: V = y1 @ S (16x512, K=64): 8 waves x 4 N-tiles ----
        {
            v8f acc[4] = {};
#pragma unroll
            for (int j = 0; j < 4; ++j) {
                const int n0 = (wave + 8 * j) * 16;
                for (int k = 0; k < NCOMBO; k += 4) {
                    v2f a, b;
                    a[0] = y1[mr * CSTR + k + kh];
                    a[1] = y1[mr * CSTR + k + kh + 1];
                    b[0] = S[(k + kh) * SSTR + n0 + mr];
                    b[1] = S[(k + kh + 1) * SSTR + n0 + mr];
                    acc[j] = __builtin_amdgcn_wmma_f32_16x16x4_f32(
                        false, a, false, b, (short)0, acc[j], false, false);
                }
            }
#pragma unroll
            for (int j = 0; j < 4; ++j) {
                const int n0 = (wave + 8 * j) * 16;
#pragma unroll
                for (int r = 0; r < 8; ++r)
                    xbV[(rowHalf + r) * SSTR + n0 + mr] = acc[j][r];
            }
        }
        __syncthreads();

        // ---- Phase D1: d = (x - tau*(V - y2)) + tau - Z ; row norms ----
        float part = 0.f;
#pragma unroll
        for (int j = 0; j < 32; ++j) {
            const int idx = drow * SSTR + dcl + 16 * j;
            float vv = x[idx] - tau * (xbV[idx] - y2[idx]);
            float d  = vv + tau - Z[idx];
            xbV[idx] = d;
            part += d * d;
        }
        red[drow * 17 + dcl] = part;
        __syncthreads();
        if (tid < 16) {
            float s = 0.f;
#pragma unroll
            for (int i = 0; i < 16; ++i) s += red[tid * 17 + i];
            float nrm = sqrtf(s);
            nrm = nrm > 1e-12f ? nrm : 1e-12f;
            float sc = 1.f - tau / nrm;
            scl[tid] = sc > 0.f ? sc : 0.f;
        }
        __syncthreads();

        // ---- Phase D2: x_new = Z + scale*d ; xbar = 2 x_new - x ----
        {
            const float sc = scl[drow];
#pragma unroll
            for (int j = 0; j < 32; ++j) {
                const int idx = drow * SSTR + dcl + 16 * j;
                float d  = xbV[idx];
                float xn = Z[idx] + sc * d;
                float xo = x[idx];
                xbV[idx] = 2.f * xn - xo;
                x[idx]   = xn;
            }
        }
        __syncthreads();
    }

    for (int e = tid; e < 16 * NSTRUCT; e += 256) {
        int r = e >> 9, c = e & 511;
        outX[(size_t)(r0 + r) * NSTRUCT + c] = x[r * SSTR + c];
    }
}

// ---------------------------------------------------------------------------
// Host launcher
// ---------------------------------------------------------------------------
extern "C" void kernel_launch(void* const* d_in, const int* in_sizes, int n_in,
                              void* d_out, int out_size, void* d_ws, size_t ws_size,
                              hipStream_t stream) {
    const float* X  = (const float*)d_in[0];   // [2048,64]
    const float* W1 = (const float*)d_in[1];   // [64,1024]
    const float* b1 = (const float*)d_in[2];
    const float* W2 = (const float*)d_in[3];   // [1024,1024]
    const float* b2 = (const float*)d_in[4];
    const float* W3 = (const float*)d_in[5];   // [1024,512]
    const float* b3 = (const float*)d_in[6];
    const float* Sf = (const float*)d_in[7];   // [64,512]
    float* out = (float*)d_out;                // [2048,512]
    (void)in_sizes; (void)n_in; (void)out_size; (void)ws_size;

    // workspace layout (16-byte aligned slabs)
    char* ws = (char*)d_ws;
    float*  tau  = (float*)ws;                               // 256 B reserved
    bf16_t* Xbf  = (bf16_t*)(ws + 256);                      // [2048][64]
    bf16_t* Wt1  = Xbf + (size_t)NB * NCOMBO;                // [1024][64]
    bf16_t* Wt2  = Wt1 + (size_t)HID * NCOMBO;               // [1024][1024]
    bf16_t* Wt3  = Wt2 + (size_t)HID * HID;                  // [512][1024]
    bf16_t* H1   = Wt3 + (size_t)NSTRUCT * HID;              // [2048][1024]
    bf16_t* H2   = H1  + (size_t)NB * HID;                   // [2048][1024]
    float*  Z    = (float*)(H2 + (size_t)NB * HID);          // [2048][512] f32

    (void)hipFuncSetAttribute(reinterpret_cast<const void*>(pdhg_kernel),
                              hipFuncAttributeMaxDynamicSharedMemorySize,
                              PDHG_LDS_BYTES);

    // 1) step size tau from power iteration on S
    power_iter_kernel<<<1, 256, 0, stream>>>(Sf, tau);

    // 2) operand prep: X -> bf16 ; W -> bf16 transposed [N][K]
    convert_f32_bf16<<<(NB * NCOMBO + 255) / 256, 256, 0, stream>>>(X, Xbf, NB * NCOMBO);
    transpose_to_bf16<<<dim3(HID / 32, NCOMBO / 32), 256, 0, stream>>>(W1, Wt1, NCOMBO, HID);
    transpose_to_bf16<<<dim3(HID / 32, HID / 32), 256, 0, stream>>>(W2, Wt2, HID, HID);
    transpose_to_bf16<<<dim3(NSTRUCT / 32, HID / 32), 256, 0, stream>>>(W3, Wt3, HID, NSTRUCT);

    // 3) MLP: relu(relu(relu(X@W1+b1)@W2+b2)@W3+b3) -> Z
    gemm_bias_relu<true ><<<dim3(HID / 128, NB / 128), 256, 0, stream>>>(
        Xbf, Wt1, b1, H1, nullptr, NB, HID, NCOMBO);
    gemm_bias_relu<true ><<<dim3(HID / 128, NB / 128), 256, 0, stream>>>(
        H1, Wt2, b2, H2, nullptr, NB, HID, HID);
    gemm_bias_relu<false><<<dim3(NSTRUCT / 128, NB / 128), 256, 0, stream>>>(
        H2, Wt3, b3, nullptr, Z, NB, NSTRUCT, HID);

    // 4) fused 60-iteration PDHG, all state LDS-resident
    pdhg_kernel<<<NB / 16, 256, PDHG_LDS_BYTES, stream>>>(Sf, Z, X, tau, out);
}
